// ResidualSTU_23321672417772
// MI455X (gfx1250) — matrix operand order repaired and checked by reading, not compiled, not run on hardware
//
#include <hip/hip_runtime.h>

// ---------------- problem constants ----------------
#define B_   4
#define SL_  4096
#define D_   128
#define K_   16
#define KU_  3
#define KY_  2
#define NT_  256           // SL_/16 time tiles
#define TSTR_ 4160         // SL_ + 64 padded stride (halfs): zero pad [0,32) and tail
#define URS_ 40            // LDS row stride (halfs) for u chunk: conflict-free, 16B aligned
#define L_   128           // scan chunk length
#define NC_  32            // SL_/L_ chunks
#define TRSG_ 136          // LDS row stride (halfs) for G matrices (conflict-free, 16B aligned)

typedef __attribute__((ext_vector_type(16))) _Float16 v16h;
typedef __attribute__((ext_vector_type(8)))  _Float16 v8h;
typedef __attribute__((ext_vector_type(8)))  float    v8f;

#define WMMA_F16(a, b, c) \
  __builtin_amdgcn_wmma_f32_16x16x32_f16(false, (a), false, (b), (short)0, (c), false, false)

#define USE_ASYNC 1

// ---------------- workspace layout (bytes) ----------------
static constexpr size_t OFF_U   = 0;                                   // uT [B][D][TSTR] fp16
static constexpr size_t SZ_U    = (size_t)B_ * D_ * TSTR_ * 2;
static constexpr size_t OFF_F   = OFF_U + SZ_U;                        // fT [2][K][TSTR] fp16
static constexpr size_t SZ_F    = (size_t)2 * K_ * TSTR_ * 2;
static constexpr size_t OFF_MPH = OFF_F + SZ_F;                        // MphT [2][K][D(o)][D(d)] fp16
static constexpr size_t SZ_MPH  = (size_t)2 * K_ * D_ * D_ * 2;
static constexpr size_t OFF_MU  = OFF_MPH + SZ_MPH;                    // MuT [KU][D(o)][D(d)] fp16
static constexpr size_t SZ_MU   = (size_t)KU_ * D_ * D_ * 2;
static constexpr size_t OFF_Y   = OFF_MU + SZ_MU;                      // y [B][SL][D] f32
static constexpr size_t SZ_Y    = (size_t)B_ * SL_ * D_ * 4;
static constexpr size_t OFF_G   = OFF_Y + SZ_Y;                        // G [L][D(o)][D(d)] fp16
static constexpr size_t SZ_G    = (size_t)L_ * D_ * D_ * 2;
static constexpr size_t OFF_V   = OFF_G + SZ_G;                        // v [B][NC][2][D] f32
static constexpr size_t SZ_V    = (size_t)B_ * NC_ * 2 * D_ * 4;

// A-fragment element -> K index map (contiguous: k = 16*lg + e, matches B loads)
__device__ __forceinline__ int kmapA(int e, int lg) { return e + 16 * lg; }

__device__ __forceinline__ v16h ld_v16h_lds(const _Float16* p) {
  union { v16h v; v8h h[2]; } u;
  u.h[0] = *(const v8h*)(p);
  u.h[1] = *(const v8h*)(p + 8);
  return u.v;
}

// ---------------- prep kernels ----------------
__global__ void prep_u_kernel(const float* __restrict__ inp, _Float16* __restrict__ uT) {
  size_t idx = (size_t)blockIdx.x * blockDim.x + threadIdx.x;
  if (idx >= (size_t)B_ * D_ * TSTR_) return;
  int tp = (int)(idx % TSTR_);
  size_t r = idx / TSTR_;
  int d = (int)(r % D_);
  int b = (int)(r / D_);
  int t = tp - 32;
  float v = 0.f;
  if (t >= 0 && t < SL_) v = inp[((size_t)b * SL_ + t) * D_ + d];
  uT[idx] = (_Float16)v;
}

__global__ void prep_f_kernel(const float* __restrict__ phi, const float* __restrict__ sigma,
                              _Float16* __restrict__ fT) {
  size_t idx = (size_t)blockIdx.x * blockDim.x + threadIdx.x;
  if (idx >= (size_t)2 * K_ * TSTR_) return;
  int tp = (int)(idx % TSTR_);
  size_t r = idx / TSTR_;
  int k = (int)(r % K_);
  int s = (int)(r / K_);
  int t = tp - 32;
  float v = 0.f;
  if (t >= 0 && t < SL_) {
    v = phi[(size_t)t * K_ + k] * sqrtf(sqrtf(sigma[k]));   // fold sigma^{1/4}
    if (s == 1 && (t & 1)) v = -v;                          // minus branch alternation
  }
  fT[idx] = (_Float16)v;
}

__global__ void prep_mph_kernel(const float* __restrict__ Mp, const float* __restrict__ Mm,
                                _Float16* __restrict__ MphT) {
  size_t idx = (size_t)blockIdx.x * blockDim.x + threadIdx.x;
  if (idx >= (size_t)2 * K_ * D_ * D_) return;
  int d = (int)(idx % D_);
  size_t r = idx / D_;
  int o = (int)(r % D_);
  size_t r2 = r / D_;
  int k = (int)(r2 % K_);
  int s = (int)(r2 / K_);
  const float* M = (s == 0) ? Mp : Mm;
  MphT[idx] = (_Float16)M[((size_t)k * D_ + d) * D_ + o];
}

__global__ void prep_mu_kernel(const float* __restrict__ Mu, _Float16* __restrict__ MuT) {
  size_t idx = (size_t)blockIdx.x * blockDim.x + threadIdx.x;
  if (idx >= (size_t)KU_ * D_ * D_) return;
  int d = (int)(idx % D_);
  size_t r = idx / D_;
  int o = (int)(r % D_);
  int i = (int)(r / D_);
  MuT[idx] = (_Float16)Mu[((size_t)i * D_ + d) * D_ + o];
}

// ---------------- spectral: LDS-shared Toeplitz WMMA conv + fused projection ----------------
__global__ __launch_bounds__(1024) void spectral_kernel(const _Float16* __restrict__ uT,
                                                        const _Float16* __restrict__ fT,
                                                        const _Float16* __restrict__ MphT,
                                                        float* __restrict__ y) {
  const int tid  = threadIdx.x;
  const int lane = tid & 31;
  const int w    = tid >> 5;
  const int ln   = lane & 15;
  const int lg   = lane >> 4;
  const int k    = w & 15;
  const int sgn  = w >> 4;
  const int I    = blockIdx.x & (NT_ - 1);
  const int b    = blockIdx.x >> 8;

  __shared__ __align__(16) _Float16 smem[24576];   // 49152 bytes
  _Float16* ubuf = smem;                // buf stride 5120 halfs (10240 B)
  _Float16* fbuf = smem + 10240;        // buf stride 2048 halfs
  _Float16* slice = smem;               // proj phase
  float*    part  = (float*)(smem + 16384);

  const int nJ = (I >> 1) + 1;
  const _Float16* ub_g = uT + (size_t)b * D_ * TSTR_;

#define STAGE(BUF, J)                                                                   \
  do {                                                                                  \
    if (tid < 512) {                                                                    \
      int row = tid >> 2, seg = tid & 3;                                                \
      const _Float16* gp = ub_g + (size_t)row * TSTR_ + 32 + 32 * (J) + seg * 8;        \
      unsigned ldsoff = (unsigned)((BUF) * 10240 + row * (URS_ * 2) + seg * 16);        \
      ASYNC_CP(ldsoff, gp);                                                             \
    } else if (tid < 768) {                                                             \
      int ftid = tid - 512;                                                             \
      int fk = ftid >> 3, seg = ftid & 7;                                               \
      const _Float16* gp = fT + (size_t)fk * TSTR_ + (16 * I - 32 * (J)) + seg * 8;     \
      *(v8h*)(fbuf + (BUF) * 2048 + fk * 64 + seg * 8) = *(const v8h*)gp;               \
    }                                                                                   \
  } while (0)

#if USE_ASYNC
#define ASYNC_CP(ldsoff, gp) \
  asm volatile("global_load_async_to_lds_b128 %0, %1, off" :: "v"(ldsoff), "v"(gp) : "memory")
#define ASYNC_WAIT() asm volatile("s_wait_asynccnt 0x0" ::: "memory")
#else
#define ASYNC_CP(ldsoff, gp) \
  *(v8h*)((_Float16*)smem + ((ldsoff) >> 1)) = *(const v8h*)(gp)
#define ASYNC_WAIT()
#endif

  v8f acc[8] = {};
  int buf = 0;
  STAGE(0, 0);
  for (int J = 0; J < nJ; ++J) {
    ASYNC_WAIT();
    __syncthreads();
    if (J + 1 < nJ) STAGE(buf ^ 1, J + 1);
    {
      const _Float16* fw = fbuf + buf * 2048 + w * 64;
      const _Float16* ub = ubuf + buf * 5120;
      v16h av;
#pragma unroll
      for (int e = 0; e < 16; ++e) av[e] = fw[32 + ln - kmapA(e, lg)];
#pragma unroll
      for (int g = 0; g < 8; ++g) {
        v16h bv = ld_v16h_lds(ub + (size_t)(g * 16 + ln) * URS_ + 16 * lg);
        acc[g] = WMMA_F16(av, bv, acc[g]);
      }
    }
    buf ^= 1;
  }
  __syncthreads();

  _Float16* mys = slice + w * (16 * 32);
  v16h ua[4];
#pragma unroll
  for (int c4 = 0; c4 < 4; ++c4) {
#pragma unroll
    for (int gg = 0; gg < 2; ++gg) {
      int g = 2 * c4 + gg;
#pragma unroll
      for (int r = 0; r < 8; ++r)
        mys[(r + 8 * lg) * 32 + gg * 16 + ln] = (_Float16)acc[g][r];
    }
#pragma unroll
    for (int e = 0; e < 16; ++e)
      ua[c4][e] = mys[ln * 32 + kmapA(e, lg)];
  }

  const _Float16* mb0 = MphT + (size_t)(sgn * K_ + k) * D_ * D_;
  float* yb = y + ((size_t)b * SL_ + 16 * I) * D_;
  for (int g2 = 0; g2 < 8; ++g2) {
    v8f s = {};
#pragma unroll
    for (int c4 = 0; c4 < 4; ++c4) {
      v16h mb = ld_v16h_lds(mb0 + (size_t)(g2 * 16 + ln) * D_ + 32 * c4 + 16 * lg);
      s = WMMA_F16(ua[c4], mb, s);
    }
    if (w >= 16) {
#pragma unroll
      for (int r = 0; r < 8; ++r) part[(w - 16) * 256 + (r + 8 * lg) * 16 + ln] = s[r];
    }
    __syncthreads();
    if (w < 16) {
#pragma unroll
      for (int r = 0; r < 8; ++r) {
        s[r] += part[w * 256 + (r + 8 * lg) * 16 + ln];
        part[w * 256 + (r + 8 * lg) * 16 + ln] = s[r];
      }
    }
    __syncthreads();
    if (tid < 256) {
      float v = 0.f;
#pragma unroll
      for (int s16 = 0; s16 < 16; ++s16) v += part[s16 * 256 + tid];
      int m = tid >> 4, n = tid & 15;
      yb[(size_t)m * D_ + g2 * 16 + n] = v;
    }
    __syncthreads();
  }
#undef STAGE
#undef ASYNC_CP
#undef ASYNC_WAIT
}

// ---------------- input AR: 3-tap matrix FIR via WMMA (accumulates into y) ----------------
__global__ void ar_u_kernel(const _Float16* __restrict__ uT,
                            const _Float16* __restrict__ MuT,
                            float* __restrict__ y) {
  const int lane = threadIdx.x;
  const int ln = lane & 15;
  const int lg = lane >> 4;
  const int I = blockIdx.x & (NT_ - 1);
  const int b = blockIdx.x >> 8;

  v8f s[8] = {};
#pragma unroll
  for (int i = 0; i < KU_; ++i) {
#pragma unroll
    for (int c4 = 0; c4 < 4; ++c4) {
      v16h ua;
#pragma unroll
      for (int e = 0; e < 16; ++e) {
        int d = 32 * c4 + kmapA(e, lg);
        ua[e] = uT[((size_t)b * D_ + d) * TSTR_ + 32 + 16 * I + ln - i];
      }
#pragma unroll
      for (int g2 = 0; g2 < 8; ++g2) {
        v16h mb = ld_v16h_lds(MuT + ((size_t)i * D_ + g2 * 16 + ln) * D_ + 32 * c4 + 16 * lg);
        s[g2] = WMMA_F16(ua, mb, s[g2]);
      }
    }
  }
  float* yb = y + ((size_t)b * SL_ + 16 * I) * D_;
#pragma unroll
  for (int g2 = 0; g2 < 8; ++g2)
#pragma unroll
    for (int r = 0; r < 8; ++r) {
      size_t off = (size_t)(r + 8 * lg) * D_ + g2 * 16 + ln;
      yb[off] = yb[off] + s[g2][r];
    }
}

// ---------------- Phase G: impulse-response matrices G_j via sequential WMMA ----------------
// G_j = M1 G_{j-1} + M2 G_{j-2}; G_{-1}=I, G_{-2}=0. One block, 32 waves, L_ steps.
// LDS (dynamic): 3 ring buffers of G, transposed [n][m], row stride TRSG_.
// Wave w: row-group mg = w>>2, col tiles ngA/ngB = (w&3)*2, +1. M1/M2 A-frags in registers.
__global__ __launch_bounds__(1024) void gmat_kernel(const float* __restrict__ My,
                                                    _Float16* __restrict__ garr) {
  extern __shared__ __align__(16) _Float16 gsm[];
  const int tid = threadIdx.x;
  const int lane = tid & 31;
  const int w = tid >> 5;
  const int ln = lane & 15;
  const int lg = lane >> 4;
  const int mg = w >> 2;
  const int ngA = (w & 3) * 2;
  const int ngB = ngA + 1;
  const int GBUF = D_ * TRSG_;   // halfs per buffer

  // init: buffers 0..2 zeroed; buffer 2 (G_{-1}) gets identity; buffer 1 = G_{-2} = 0
  for (int i = tid; i < 3 * GBUF; i += 1024) gsm[i] = (_Float16)0.f;
  __syncthreads();
  if (tid < D_) gsm[2 * GBUF + tid * TRSG_ + tid] = (_Float16)1.f;

  // preload A-fragments of M1 (My[:,0,:]) and M2 (My[:,1,:])
  v16h a1[4], a2[4];
#pragma unroll
  for (int c4 = 0; c4 < 4; ++c4)
#pragma unroll
    for (int e = 0; e < 16; ++e) {
      int d = 32 * c4 + kmapA(e, lg);
      int o = mg * 16 + ln;
      a1[c4][e] = (_Float16)My[((size_t)o * KY_ + 0) * D_ + d];
      a2[c4][e] = (_Float16)My[((size_t)o * KY_ + 1) * D_ + d];
    }
  __syncthreads();

  for (int j = 0; j < L_; ++j) {
    const _Float16* g1 = gsm + (size_t)((j + 2) % 3) * GBUF;   // G_{j-1}
    const _Float16* g2 = gsm + (size_t)((j + 1) % 3) * GBUF;   // G_{j-2}
    _Float16*       go = gsm + (size_t)(j % 3) * GBUF;         // G_j
    v8f accA = {}, accB = {};
#pragma unroll
    for (int c4 = 0; c4 < 4; ++c4) {
      v16h bA1 = ld_v16h_lds(g1 + (size_t)(ngA * 16 + ln) * TRSG_ + 32 * c4 + 16 * lg);
      accA = WMMA_F16(a1[c4], bA1, accA);
      v16h bA2 = ld_v16h_lds(g2 + (size_t)(ngA * 16 + ln) * TRSG_ + 32 * c4 + 16 * lg);
      accA = WMMA_F16(a2[c4], bA2, accA);
      v16h bB1 = ld_v16h_lds(g1 + (size_t)(ngB * 16 + ln) * TRSG_ + 32 * c4 + 16 * lg);
      accB = WMMA_F16(a1[c4], bB1, accB);
      v16h bB2 = ld_v16h_lds(g2 + (size_t)(ngB * 16 + ln) * TRSG_ + 32 * c4 + 16 * lg);
      accB = WMMA_F16(a2[c4], bB2, accB);
    }
    v8h hA, hB;
#pragma unroll
    for (int r = 0; r < 8; ++r) { hA[r] = (_Float16)accA[r]; hB[r] = (_Float16)accB[r]; }
    // transposed LDS store: [n][m], m-run of 8 contiguous
    *(v8h*)(go + (size_t)(ngA * 16 + ln) * TRSG_ + mg * 16 + 8 * lg) = hA;
    *(v8h*)(go + (size_t)(ngB * 16 + ln) * TRSG_ + mg * 16 + 8 * lg) = hB;
    // global row-major store [j][o][d]
    _Float16* gg = garr + (size_t)j * D_ * D_;
#pragma unroll
    for (int r = 0; r < 8; ++r) {
      gg[(size_t)(mg * 16 + r + 8 * lg) * D_ + ngA * 16 + ln] = hA[r];
      gg[(size_t)(mg * 16 + r + 8 * lg) * D_ + ngB * 16 + ln] = hB[r];
    }
    __syncthreads();
  }
}

// ---------------- Phase A: zero-state scan within each chunk (parallel over chunks) -------
__global__ void scanA_kernel(const float* __restrict__ y, const float* __restrict__ My,
                             float* __restrict__ out) {
  const int c = blockIdx.x & (NC_ - 1);
  const int b = blockIdx.x >> 5;
  const int tid = threadIdx.x;
  const int q = tid >> 7;
  const int o = tid & 127;
  const int hh = q >> 1;
  const int ds = (q & 1) * 64;

  __shared__ float hbuf[2][D_];
  __shared__ float part[512];

  float w[64];
#pragma unroll
  for (int j = 0; j < 64; ++j) w[j] = My[((size_t)o * KY_ + hh) * D_ + ds + j];

  if (tid < 2 * D_) ((float*)hbuf)[tid] = 0.f;
  __syncthreads();

  const float* yb = y + ((size_t)b * SL_ + (size_t)c * L_) * D_;
  float* ob = out + ((size_t)b * SL_ + (size_t)c * L_) * D_;

  for (int t = 0; t < L_; ++t) {
    const int cur = t & 1;
    if (q == 0 && (t & 15) == 0 && t + 16 < L_)
      __builtin_prefetch(&yb[(size_t)(t + 16) * D_ + o], 0, 0);
    const float* hp = (hh == 0) ? hbuf[1 - cur] : hbuf[cur];
    float p = 0.f;
#pragma unroll
    for (int j = 0; j < 64; ++j) p += w[j] * hp[ds + j];
    part[tid] = p;
    __syncthreads();
    if (q == 0) {
      float v = yb[(size_t)t * D_ + o] + part[o] + part[128 + o] + part[256 + o] + part[384 + o];
      ob[(size_t)t * D_ + o] = v;
      hbuf[cur][o] = v;
    }
    __syncthreads();
  }
}

// ---------------- Phase B: sequential chunk-boundary propagation ----------------
// Per batch b: h1,h2 = 0; for each chunk c: v1=h1, v2=M2 h2 (saved), then
// e1 = p_{L-1} + G_{L-1} v1 + G_{L-2} v2 ; e2 = p_{L-2} + G_{L-2} v1 + G_{L-3} v2.
__global__ void chainB_kernel(const float* __restrict__ My, const _Float16* __restrict__ garr,
                              const float* __restrict__ out, float* __restrict__ varr) {
  const int b = blockIdx.x;
  const int o = threadIdx.x;   // 128 threads
  __shared__ float h1[D_], h2[D_], v2s[D_];

  h1[o] = 0.f; h2[o] = 0.f;
  __syncthreads();

  const _Float16* GL1 = garr + (size_t)(L_ - 1) * D_ * D_;
  const _Float16* GL2 = garr + (size_t)(L_ - 2) * D_ * D_;
  const _Float16* GL3 = garr + (size_t)(L_ - 3) * D_ * D_;

  for (int c = 0; c < NC_; ++c) {
    // v2 = M2 h2
    float a = 0.f;
    for (int d = 0; d < D_; ++d) a += My[((size_t)o * KY_ + 1) * D_ + d] * h2[d];
    // save v1 (= h1) and v2
    float* vb = varr + (size_t)((b * NC_ + c) * 2) * D_;
    vb[o] = h1[o];
    vb[D_ + o] = a;
    v2s[o] = a;
    __syncthreads();
    // end states of this chunk
    float e1 = out[((size_t)b * SL_ + (size_t)c * L_ + (L_ - 1)) * D_ + o];
    float e2 = out[((size_t)b * SL_ + (size_t)c * L_ + (L_ - 2)) * D_ + o];
    for (int d = 0; d < D_; ++d) {
      float x1 = h1[d], x2 = v2s[d];
      e1 += (float)GL1[(size_t)o * D_ + d] * x1 + (float)GL2[(size_t)o * D_ + d] * x2;
      e2 += (float)GL2[(size_t)o * D_ + d] * x1 + (float)GL3[(size_t)o * D_ + d] * x2;
    }
    __syncthreads();
    h1[o] = e1; h2[o] = e2;
    __syncthreads();
  }
}

// ---------------- Phase C: parallel rank-2 correction ----------------
// out[b, c*L+j, o] += G_j[o,:] v1[b,c] + (j>0 ? G_{j-1}[o,:] v2[b,c] : v2[b,c][o])
__global__ void corrC_kernel(const _Float16* __restrict__ garr, const float* __restrict__ varr,
                             float* __restrict__ out) {
  const int win = blockIdx.x & 127;       // 32-step window
  const int b = blockIdx.x >> 7;
  const int tid = threadIdx.x;            // 256
  const int t0 = win * 32;
  const int c = t0 >> 7;                  // chunk (L_=128)
  const int j0 = t0 & (L_ - 1);

  __shared__ float v1[D_], v2[D_];
  if (tid < D_) v1[tid] = varr[(size_t)((b * NC_ + c) * 2) * D_ + tid];
  else v2[tid - D_] = varr[(size_t)((b * NC_ + c) * 2) * D_ + (tid - D_) + D_];
  __syncthreads();

  const int tl = tid >> 3;                // 0..31 time step in window
  const int j = j0 + tl;
  const int t = t0 + tl;
  float* ob = out + ((size_t)b * SL_ + t) * D_;
  const v8h* Gj = (const v8h*)(garr + (size_t)j * D_ * D_);
  const v8h* Gp = (const v8h*)(garr + (size_t)(j > 0 ? j - 1 : 0) * D_ * D_);

#pragma unroll
  for (int oo = 0; oo < 16; ++oo) {
    int o = (tid & 7) * 16 + oo;
    float acc = 0.f;
    const v8h* gr = Gj + (size_t)o * 16;
#pragma unroll 4
    for (int dv = 0; dv < 16; ++dv) {
      v8h g = gr[dv];
#pragma unroll
      for (int e = 0; e < 8; ++e) acc += (float)g[e] * v1[dv * 8 + e];
    }
    if (j > 0) {
      const v8h* gp = Gp + (size_t)o * 16;
#pragma unroll 4
      for (int dv = 0; dv < 16; ++dv) {
        v8h g = gp[dv];
#pragma unroll
        for (int e = 0; e < 8; ++e) acc += (float)g[e] * v2[dv * 8 + e];
      }
    } else {
      acc += v2[o];
    }
    ob[o] += acc;
  }
}

// ---------------- launch ----------------
extern "C" void kernel_launch(void* const* d_in, const int* in_sizes, int n_in,
                              void* d_out, int out_size, void* d_ws, size_t ws_size,
                              hipStream_t stream) {
  const float* inputs = (const float*)d_in[0];
  const float* phi    = (const float*)d_in[1];
  const float* sigma  = (const float*)d_in[2];
  const float* Mu     = (const float*)d_in[3];
  const float* Mp     = (const float*)d_in[4];
  const float* Mm     = (const float*)d_in[5];
  const float* My     = (const float*)d_in[6];
  float* out = (float*)d_out;

  char* ws = (char*)d_ws;
  _Float16* uT   = (_Float16*)(ws + OFF_U);
  _Float16* fT   = (_Float16*)(ws + OFF_F);
  _Float16* MphT = (_Float16*)(ws + OFF_MPH);
  _Float16* MuT  = (_Float16*)(ws + OFF_MU);
  float*    y    = (float*)(ws + OFF_Y);
  _Float16* garr = (_Float16*)(ws + OFF_G);
  float*    varr = (float*)(ws + OFF_V);

  {
    size_t n = (size_t)B_ * D_ * TSTR_;
    prep_u_kernel<<<dim3((unsigned)((n + 255) / 256)), dim3(256), 0, stream>>>(inputs, uT);
  }
  {
    size_t n = (size_t)2 * K_ * TSTR_;
    prep_f_kernel<<<dim3((unsigned)((n + 255) / 256)), dim3(256), 0, stream>>>(phi, sigma, fT);
  }
  {
    size_t n = (size_t)2 * K_ * D_ * D_;
    prep_mph_kernel<<<dim3((unsigned)((n + 255) / 256)), dim3(256), 0, stream>>>(Mp, Mm, MphT);
  }
  {
    size_t n = (size_t)KU_ * D_ * D_;
    prep_mu_kernel<<<dim3((unsigned)((n + 255) / 256)), dim3(256), 0, stream>>>(Mu, MuT);
  }

  spectral_kernel<<<dim3(B_ * NT_), dim3(1024), 0, stream>>>(uT, fT, MphT, y);
  ar_u_kernel<<<dim3(B_ * NT_), dim3(32), 0, stream>>>(uT, MuT, y);

  // chunk-parallel linear recurrence
  gmat_kernel<<<dim3(1), dim3(1024), (size_t)(3 * D_ * TRSG_ * 2), stream>>>(My, garr);
  scanA_kernel<<<dim3(B_ * NC_), dim3(512), 0, stream>>>(y, My, out);
  chainB_kernel<<<dim3(B_), dim3(D_), 0, stream>>>(My, garr, out, varr);
  corrC_kernel<<<dim3(B_ * 128), dim3(256), 0, stream>>>(garr, varr, out);
}
